// Multi_Head_Attention_25958782337285
// MI455X (gfx1250) — 16-bit and 8-bit matrix operand order repaired and checked by reading, compiled, not run on hardware
//
#include <hip/hip_runtime.h>

typedef __attribute__((ext_vector_type(16))) __bf16 v16bf;
typedef __attribute__((ext_vector_type(8)))  __bf16 v8bf;
typedef __attribute__((ext_vector_type(8)))  float  v8f;
typedef __attribute__((ext_vector_type(8)))  int    v8i;
typedef __attribute__((ext_vector_type(4)))  int    v4i;
typedef __attribute__((ext_vector_type(2)))  int    v2i;

#define B_  8
#define S_  1024
#define D_  512
#define H_  8
#define DK_ 64

#if __has_builtin(__builtin_amdgcn_global_load_async_to_lds_b128)
#define HAVE_ASYNC_LDS 1
#else
#define HAVE_ASYNC_LDS 0
#endif

__device__ __forceinline__ void async_copy16(const void* g, void* l) {
#if HAVE_ASYNC_LDS
  __builtin_amdgcn_global_load_async_to_lds_b128((v4i*)g, (v4i*)l, 0, 0);
#else
  *(v4i*)l = *(const v4i*)g;
#endif
}
__device__ __forceinline__ void wait_async() {
#if HAVE_ASYNC_LDS
#if __has_builtin(__builtin_amdgcn_s_wait_asynccnt)
  __builtin_amdgcn_s_wait_asynccnt(0);
#else
  asm volatile("s_wait_asynccnt 0x0" ::: "memory");
#endif
#endif
}

__device__ __forceinline__ v8f wmma_bf16(v16bf a, v16bf b, v8f c) {
  return __builtin_amdgcn_wmma_f32_16x16x32_bf16(false, a, false, b,
                                                 (short)0, c, false, false);
}
__device__ __forceinline__ v8i wmma_u8(v8i a, v8i b, v8i c) {
  return __builtin_amdgcn_wmma_i32_16x16x64_iu8(false, a, false, b, c, false, false);
}

__device__ __forceinline__ v16bf load_a_bf16(const __bf16* base, int ld) {
  const int lane = threadIdx.x & 31;
  const int row  = lane & 15;
  const int half = lane >> 4;
  const __bf16* p = base + row * ld + half * 8;
  v8bf lo = *(const v8bf*)(p);
  v8bf hi = *(const v8bf*)(p + 16);
  return __builtin_shufflevector(lo, hi, 0,1,2,3,4,5,6,7,8,9,10,11,12,13,14,15);
}
__device__ __forceinline__ v16bf load_b_bf16(const __bf16* bt, int ld) {
  const int lane = threadIdx.x & 31;
  const int n    = lane & 15;
  const int half = lane >> 4;
  const __bf16* p = bt + n * ld + half * 8;
  v8bf lo = *(const v8bf*)(p);
  v8bf hi = *(const v8bf*)(p + 16);
  return __builtin_shufflevector(lo, hi, 0,1,2,3,4,5,6,7,8,9,10,11,12,13,14,15);
}
__device__ __forceinline__ v8i load_a_u8(const unsigned char* base, int ld) {
  const int lane = threadIdx.x & 31;
  const int row  = lane & 15;
  const int half = lane >> 4;
  const unsigned char* p = base + row * ld + half * 8;
  v2i c0 = *(const v2i*)(p);
  v2i c1 = *(const v2i*)(p + 16);
  v2i c2 = *(const v2i*)(p + 32);
  v2i c3 = *(const v2i*)(p + 48);
  v8i r;
  r[0]=c0[0]; r[1]=c0[1]; r[2]=c1[0]; r[3]=c1[1];
  r[4]=c2[0]; r[5]=c2[1]; r[6]=c3[0]; r[7]=c3[1];
  return r;
}
__device__ __forceinline__ v8i load_b_u8(const unsigned char* bt, int ld) {
  const int lane = threadIdx.x & 31;
  const int n    = lane & 15;
  const int half = lane >> 4;
  const unsigned char* p = bt + n * ld + half * 8;
  v2i c0 = *(const v2i*)(p);
  v2i c1 = *(const v2i*)(p + 16);
  v2i c2 = *(const v2i*)(p + 32);
  v2i c3 = *(const v2i*)(p + 48);
  v8i r;
  r[0]=c0[0]; r[1]=c0[1]; r[2]=c1[0]; r[3]=c1[1];
  r[4]=c2[0]; r[5]=c2[1]; r[6]=c3[0]; r[7]=c3[1];
  return r;
}

__global__ __launch_bounds__(256) void qkv_proj_kernel(
    const float* __restrict__ X,
    const float* __restrict__ Wq, const float* __restrict__ Wk, const float* __restrict__ Wv,
    const float* __restrict__ bq, const float* __restrict__ bk, const float* __restrict__ bv,
    __bf16* __restrict__ Qo, __bf16* __restrict__ Ko, __bf16* __restrict__ Vo)
{
  const int which = blockIdx.z;
  const float* W    = (which == 0) ? Wq : (which == 1) ? Wk : Wv;
  const float* bias = (which == 0) ? bq : (which == 1) ? bk : bv;
  __bf16*      Out  = (which == 0) ? Qo : (which == 1) ? Ko : Vo;

  __shared__ __bf16 la [128 * 32];
  __shared__ __bf16 lbt[128 * 32];

  const int tid  = threadIdx.x;
  const int m0   = blockIdx.y * 128;
  const int n0   = blockIdx.x * 128;
  const int wave = tid >> 5;
  const int wm   = (wave & 3) * 32;
  const int wn   = (wave >> 2) * 64;

  v8f acc[2][4] = {};

  for (int k0 = 0; k0 < D_; k0 += 32) {
    __syncthreads();
    #pragma unroll
    for (int i = 0; i < 16; ++i) {
      int idx = i * 256 + tid;
      int m = idx >> 5, k = idx & 31;
      la[idx] = (__bf16)X[(m0 + m) * D_ + k0 + k];
    }
    #pragma unroll
    for (int i = 0; i < 16; ++i) {
      int idx = i * 256 + tid;
      int k = idx >> 7, n = idx & 127;
      lbt[n * 32 + k] = (__bf16)W[(k0 + k) * D_ + n0 + n];
    }
    __syncthreads();

    v16bf a0 = load_a_bf16(la + (wm +  0) * 32, 32);
    v16bf a1 = load_a_bf16(la + (wm + 16) * 32, 32);
    #pragma unroll
    for (int ni = 0; ni < 4; ++ni) {
      v16bf bf = load_b_bf16(lbt + (wn + ni * 16) * 32, 32);
      acc[0][ni] = wmma_bf16(a0, bf, acc[0][ni]);
      acc[1][ni] = wmma_bf16(a1, bf, acc[1][ni]);
    }
  }

  const int lane = tid & 31;
  const int col  = lane & 15;
  const int rb   = (lane >> 4) * 8;
  #pragma unroll
  for (int mi = 0; mi < 2; ++mi)
    #pragma unroll
    for (int ni = 0; ni < 4; ++ni) {
      int n_g = n0 + wn + ni * 16 + col;
      float bb = bias[n_g];
      int h = n_g >> 6, dk = n_g & 63;
      #pragma unroll
      for (int r = 0; r < 8; ++r) {
        int m_g = m0 + wm + mi * 16 + rb + r;
        int b = m_g >> 10, s = m_g & 1023;
        Out[(((b * H_ + h) * S_) + s) * DK_ + dk] = (__bf16)(acc[mi][ni][r] + bb);
      }
    }
}

__global__ __launch_bounds__(256) void node_scores_kernel(
    const float* __restrict__ NK, float* __restrict__ NS)
{
  const int b  = blockIdx.z;
  const int m0 = blockIdx.y * 128;
  const int n0 = blockIdx.x * 128;

  __shared__ __bf16 la[128 * 64];
  __shared__ __bf16 lb[128 * 64];

  const int tid  = threadIdx.x;
  const int wave = tid >> 5;
  const int wm   = (wave & 3) * 32;
  const int wn   = (wave >> 2) * 64;
  const float* nk = NK + (size_t)b * S_ * DK_;

  #pragma unroll
  for (int i = 0; i < 32; ++i) {
    int idx = i * 256 + tid;
    int r = idx >> 6, d = idx & 63;
    la[idx] = (__bf16)nk[(m0 + r) * DK_ + d];
    lb[idx] = (__bf16)nk[(n0 + r) * DK_ + d];
  }
  __syncthreads();

  v8f acc[2][4] = {};
  #pragma unroll
  for (int ks = 0; ks < 64; ks += 32) {
    v16bf a0 = load_a_bf16(la + (wm +  0) * 64 + ks, 64);
    v16bf a1 = load_a_bf16(la + (wm + 16) * 64 + ks, 64);
    #pragma unroll
    for (int ni = 0; ni < 4; ++ni) {
      v16bf bf = load_b_bf16(lb + (wn + ni * 16) * 64 + ks, 64);
      acc[0][ni] = wmma_bf16(a0, bf, acc[0][ni]);
      acc[1][ni] = wmma_bf16(a1, bf, acc[1][ni]);
    }
  }

  const int lane = tid & 31, col = lane & 15, rb = (lane >> 4) * 8;
  float* dst = NS + (size_t)b * S_ * S_;
  #pragma unroll
  for (int mi = 0; mi < 2; ++mi)
    #pragma unroll
    for (int ni = 0; ni < 4; ++ni) {
      int n_g = n0 + wn + ni * 16 + col;
      #pragma unroll
      for (int r = 0; r < 8; ++r) {
        int m_g = m0 + wm + mi * 16 + rb + r;
        dst[(size_t)m_g * S_ + n_g] = acc[mi][ni][r] * 0.125f;
      }
    }
}

__global__ __launch_bounds__(256) void adj_att_kernel(
    const float* __restrict__ ADJ, float* __restrict__ PA)
{
  const int b  = blockIdx.z;
  const int m0 = blockIdx.y * 128;
  const int n0 = blockIdx.x * 128;

  __shared__ unsigned char la[128 * 64];
  __shared__ unsigned char lb[128 * 64];

  const int tid  = threadIdx.x;
  const int wave = tid >> 5;
  const int wm   = (wave & 3) * 32;
  const int wn   = (wave >> 2) * 64;
  const float* A = ADJ + (size_t)b * S_ * S_;

  v8i acc[2][4] = {};

  for (int k0 = 0; k0 < S_; k0 += 64) {
    __syncthreads();
    #pragma unroll
    for (int i = 0; i < 32; ++i) {
      int idx = i * 256 + tid;
      int r = idx >> 6, k = idx & 63;
      la[idx] = (unsigned char)A[(size_t)(m0 + r) * S_ + k0 + k];
      lb[idx] = (unsigned char)A[(size_t)(n0 + r) * S_ + k0 + k];
    }
    __syncthreads();

    v8i a0 = load_a_u8(la + (wm +  0) * 64, 64);
    v8i a1 = load_a_u8(la + (wm + 16) * 64, 64);
    #pragma unroll
    for (int ni = 0; ni < 4; ++ni) {
      v8i bf = load_b_u8(lb + (wn + ni * 16) * 64, 64);
      acc[0][ni] = wmma_u8(a0, bf, acc[0][ni]);
      acc[1][ni] = wmma_u8(a1, bf, acc[1][ni]);
    }
  }

  const int lane = tid & 31, col = lane & 15, rb = (lane >> 4) * 8;
  float* dst = PA + (size_t)b * S_ * S_;
  #pragma unroll
  for (int mi = 0; mi < 2; ++mi)
    #pragma unroll
    for (int ni = 0; ni < 4; ++ni) {
      int n_g = n0 + wn + ni * 16 + col;
      #pragma unroll
      for (int r = 0; r < 8; ++r) {
        int m_g = m0 + wm + mi * 16 + rb + r;
        dst[(size_t)m_g * S_ + n_g] = (float)acc[mi][ni][r];
      }
    }
}

__global__ __launch_bounds__(256) void flash_attn_kernel(
    const __bf16* __restrict__ Q, const __bf16* __restrict__ K,
    const __bf16* __restrict__ V,
    const float* __restrict__ PA, const float* __restrict__ NS,
    const float* __restrict__ BIAS, __bf16* __restrict__ CTX)
{
  const int s0   = blockIdx.x * 128;
  const int h    = blockIdx.y;
  const int b    = blockIdx.z;
  const int tid  = threadIdx.x;
  const int wave = tid >> 5;
  const int lane = tid & 31;
  const int col  = lane & 15;
  const int rb   = (lane >> 4) * 8;
  const int srow = s0 + wave * 16;

  __shared__ __bf16 lds_k [128 * 64];
  __shared__ __bf16 lds_vt[64 * 128];
  __shared__ __bf16 lds_p [8 * 16 * 128];

  const size_t head = ((size_t)b * H_ + h) * S_;
  const __bf16* q  = Q + head * DK_;
  const __bf16* kk = K + head * DK_;
  const __bf16* vv = V + head * DK_;
  const float*  pa   = PA   + (size_t)b * S_ * S_;
  const float*  ns   = NS   + (size_t)b * S_ * S_;
  const float*  bias = BIAS + head * S_;
  __bf16* pw = lds_p + wave * 16 * 128;

  v16bf aq0 = load_a_bf16(q + (size_t)srow * DK_ +  0, DK_);
  v16bf aq1 = load_a_bf16(q + (size_t)srow * DK_ + 32, DK_);

  float mrow[8], lrow[8];
  #pragma unroll
  for (int r = 0; r < 8; ++r) { mrow[r] = -1e30f; lrow[r] = 0.f; }
  v8f oacc[4] = {};
  const float scale = 0.125f;

  for (int t0 = 0; t0 < S_; t0 += 128) {
    __syncthreads();
    #pragma unroll
    for (int i = 0; i < 4; ++i) {
      int idx = (i * 256 + tid) * 8;
      async_copy16(kk + (size_t)t0 * DK_ + idx, lds_k + idx);
    }
    #pragma unroll
    for (int i = 0; i < 32; ++i) {
      int idx = i * 256 + tid;
      int t = idx >> 6, d = idx & 63;
      lds_vt[d * 128 + t] = vv[(size_t)(t0 + t) * DK_ + d];
    }
    if (t0 + 128 < S_) {
      #pragma unroll
      for (int r = 0; r < 8; r += 2)
        __builtin_prefetch(bias + (size_t)(srow + rb + r) * S_ + t0 + 128 + col, 0, 1);
    }
    wait_async();
    __syncthreads();

    v8f sacc[8];
    #pragma unroll
    for (int nt = 0; nt < 8; ++nt) {
      v8f z = {};
      v16bf b0 = load_b_bf16(lds_k + nt * 16 * 64 +  0, 64);
      v16bf b1 = load_b_bf16(lds_k + nt * 16 * 64 + 32, 64);
      z = wmma_bf16(aq0, b0, z);
      z = wmma_bf16(aq1, b1, z);
      sacc[nt] = z;
    }

    #pragma unroll
    for (int nt = 0; nt < 8; ++nt) {
      int t_g = t0 + nt * 16 + col;
      #pragma unroll
      for (int r = 0; r < 8; ++r) {
        size_t off = (size_t)(srow + rb + r) * S_ + t_g;
        sacc[nt][r] = sacc[nt][r] * scale * pa[off] + ns[off] + bias[off];
      }
    }

    float tmax[8];
    #pragma unroll
    for (int r = 0; r < 8; ++r) {
      float m = sacc[0][r];
      #pragma unroll
      for (int nt = 1; nt < 8; ++nt) m = fmaxf(m, sacc[nt][r]);
      tmax[r] = m;
    }
    #pragma unroll
    for (int off = 1; off < 16; off <<= 1)
      #pragma unroll
      for (int r = 0; r < 8; ++r)
        tmax[r] = fmaxf(tmax[r], __shfl_xor(tmax[r], off, 32));

    float alpha[8];
    #pragma unroll
    for (int r = 0; r < 8; ++r) {
      float mnew = fmaxf(mrow[r], tmax[r]);
      alpha[r] = __expf(mrow[r] - mnew);
      mrow[r] = mnew;
    }

    float tsum[8];
    #pragma unroll
    for (int r = 0; r < 8; ++r) tsum[r] = 0.f;
    #pragma unroll
    for (int nt = 0; nt < 8; ++nt)
      #pragma unroll
      for (int r = 0; r < 8; ++r) {
        float p = __expf(sacc[nt][r] - mrow[r]);
        sacc[nt][r] = p;
        tsum[r] += p;
      }
    #pragma unroll
    for (int off = 1; off < 16; off <<= 1)
      #pragma unroll
      for (int r = 0; r < 8; ++r)
        tsum[r] += __shfl_xor(tsum[r], off, 32);
    #pragma unroll
    for (int r = 0; r < 8; ++r) lrow[r] = lrow[r] * alpha[r] + tsum[r];

    #pragma unroll
    for (int nd = 0; nd < 4; ++nd)
      #pragma unroll
      for (int r = 0; r < 8; ++r) oacc[nd][r] *= alpha[r];

    #pragma unroll
    for (int nt = 0; nt < 8; ++nt)
      #pragma unroll
      for (int r = 0; r < 8; ++r)
        pw[(rb + r) * 128 + nt * 16 + col] = (__bf16)sacc[nt][r];

    #pragma unroll
    for (int kt = 0; kt < 4; ++kt) {
      v16bf ap = load_a_bf16(pw + kt * 32, 128);
      #pragma unroll
      for (int nd = 0; nd < 4; ++nd) {
        v16bf bv_ = load_b_bf16(lds_vt + nd * 16 * 128 + kt * 32, 128);
        oacc[nd] = wmma_bf16(ap, bv_, oacc[nd]);
      }
    }
  }

  #pragma unroll
  for (int r = 0; r < 8; ++r) {
    float inv = 1.0f / (lrow[r] + 1e-20f);
    int s_g = srow + rb + r;
    #pragma unroll
    for (int nd = 0; nd < 4; ++nd) {
      int d_g = nd * 16 + col;
      CTX[((size_t)(b * S_ + s_g)) * D_ + h * DK_ + d_g] = (__bf16)(oacc[nd][r] * inv);
    }
  }
}

__global__ __launch_bounds__(256) void out_proj_kernel(
    const __bf16* __restrict__ CTX, const float* __restrict__ Wo,
    const float* __restrict__ bo, float* __restrict__ OUT)
{
  __shared__ __bf16 la [128 * 32];
  __shared__ __bf16 lbt[128 * 32];

  const int tid  = threadIdx.x;
  const int m0   = blockIdx.y * 128;
  const int n0   = blockIdx.x * 128;
  const int wave = tid >> 5;
  const int wm   = (wave & 3) * 32;
  const int wn   = (wave >> 2) * 64;

  v8f acc[2][4] = {};

  for (int k0 = 0; k0 < D_; k0 += 32) {
    __syncthreads();
    #pragma unroll
    for (int i = 0; i < 2; ++i) {
      int idx = (i * 256 + tid) * 8;
      int m = idx >> 5, k = idx & 31;
      async_copy16(CTX + (size_t)(m0 + m) * D_ + k0 + k, la + idx);
    }
    #pragma unroll
    for (int i = 0; i < 16; ++i) {
      int idx = i * 256 + tid;
      int k = idx >> 7, n = idx & 127;
      lbt[n * 32 + k] = (__bf16)Wo[(size_t)(k0 + k) * D_ + n0 + n];
    }
    wait_async();
    __syncthreads();

    v16bf a0 = load_a_bf16(la + (wm +  0) * 32, 32);
    v16bf a1 = load_a_bf16(la + (wm + 16) * 32, 32);
    #pragma unroll
    for (int ni = 0; ni < 4; ++ni) {
      v16bf bf = load_b_bf16(lbt + (wn + ni * 16) * 32, 32);
      acc[0][ni] = wmma_bf16(a0, bf, acc[0][ni]);
      acc[1][ni] = wmma_bf16(a1, bf, acc[1][ni]);
    }
  }

  const int lane = tid & 31, col = lane & 15, rb = (lane >> 4) * 8;
  #pragma unroll
  for (int mi = 0; mi < 2; ++mi)
    #pragma unroll
    for (int ni = 0; ni < 4; ++ni) {
      int n_g = n0 + wn + ni * 16 + col;
      float bb = bo[n_g];
      #pragma unroll
      for (int r = 0; r < 8; ++r) {
        int m_g = m0 + wm + mi * 16 + rb + r;
        OUT[(size_t)m_g * D_ + n_g] = acc[mi][ni][r] + bb;
      }
    }
}

extern "C" void kernel_launch(void* const* d_in, const int* in_sizes, int n_in,
                              void* d_out, int out_size, void* d_ws, size_t ws_size,
                              hipStream_t stream)
{
  (void)in_sizes; (void)n_in; (void)out_size; (void)ws_size;
  const float* queries = (const float*)d_in[0];
  const float* adj     = (const float*)d_in[1];
  const float* nkey    = (const float*)d_in[2];
  const float* bias    = (const float*)d_in[4];
  const float* Wq = (const float*)d_in[5];
  const float* bq = (const float*)d_in[6];
  const float* Wk = (const float*)d_in[7];
  const float* bk = (const float*)d_in[8];
  const float* Wv = (const float*)d_in[9];
  const float* bv = (const float*)d_in[10];
  const float* Wo = (const float*)d_in[11];
  const float* bo = (const float*)d_in[12];
  float* out = (float*)d_out;

  char* ws = (char*)d_ws;
  __bf16* Qb  = (__bf16*)(ws + (size_t)0);
  __bf16* Kb  = (__bf16*)(ws + ((size_t)8  << 20));
  __bf16* Vb  = (__bf16*)(ws + ((size_t)16 << 20));
  __bf16* CTX = (__bf16*)(ws + ((size_t)24 << 20));
  float*  PA  = (float*) (ws + ((size_t)32 << 20));
  float*  NS  = (float*) (ws + ((size_t)64 << 20));

  qkv_proj_kernel   <<<dim3(4, 64, 3), 256, 0, stream>>>(queries, Wq, Wk, Wv, bq, bk, bv, Qb, Kb, Vb);
  node_scores_kernel<<<dim3(8, 8, 8),  256, 0, stream>>>(nkey, NS);
  adj_att_kernel    <<<dim3(8, 8, 8),  256, 0, stream>>>(adj, PA);
  flash_attn_kernel <<<dim3(8, 8, 8),  256, 0, stream>>>(Qb, Kb, Vb, PA, NS, bias, CTX);
  out_proj_kernel   <<<dim3(4, 64, 1), 256, 0, stream>>>(CTX, Wo, bo, out);
}
